// Mamba_18511309045852
// MI455X (gfx1250) — matrix operand order repaired and checked by reading, not compiled, not run on hardware
//
#include <hip/hip_runtime.h>
#include <math.h>

// ---------------- problem constants ----------------
#define BB    2
#define LL    2048
#define DMOD  2048
#define DI_   4096
#define DXB_  1024
#define NST   16
#define DTR_  128
#define KCV   4
#define H_    256          // DI/N
#define REP_  4            // H / HXB
#define ZW    (2*DXB_ + 2*DI_)   // 10240 columns of zxbc
#define MROWS (BB*LL)            // 4096

typedef __bf16 v16bf __attribute__((ext_vector_type(16)));
typedef __bf16 v8bf  __attribute__((ext_vector_type(8)));
typedef float  v8f   __attribute__((ext_vector_type(8)));

__device__ __forceinline__ unsigned short f2bf(float f) {
  unsigned int u = __float_as_uint(f);
  u += 0x7FFFu + ((u >> 16) & 1u);        // round-to-nearest-even
  return (unsigned short)(u >> 16);
}

// ---------------- fp32 -> bf16 convert ----------------
__global__ void cvt_f32_bf16(const float* __restrict__ src,
                             unsigned short* __restrict__ dst, int n) {
  int i = blockIdx.x * blockDim.x + threadIdx.x;
  if (i < n) dst[i] = f2bf(src[i]);
}

// ---------------- WMMA GEMM: C[M,Nc] = A[M,K] * W[Nc,K]^T ----------------
// Wave tile: 32(M) x 64(N). 2 A-frags x 4 B-frags -> 8 wmma per 32-K step.
// Two-stage software pipeline (ping/pong fragment buffers); K % 64 == 0.

__device__ __forceinline__ v16bf load_afrag(const __bf16* aRow, int k0, int hlf) {
  // A (16x32 bf16): lane = {hlf, m}; halves[0..7]=K[k0+8h..], [8..15]=K[k0+16+8h..]
  union { v16bf f; v8bf h[2]; } ua;
  ua.h[0] = *(const v8bf*)(aRow + k0 + 8 * hlf);
  ua.h[1] = *(const v8bf*)(aRow + k0 + 16 + 8 * hlf);
  return ua.f;
}
__device__ __forceinline__ v16bf load_bfrag(const __bf16* bRow, int k0, int hlf) {
  // B (32x16 bf16): lane holds column r = W row; halves[0..7]=K[k0+8h..], [8..15]=K[k0+16+8h..]
  union { v16bf f; v8bf h[2]; } ub;
  ub.h[0] = *(const v8bf*)(bRow + k0 + 8 * hlf);
  ub.h[1] = *(const v8bf*)(bRow + k0 + 16 + 8 * hlf);
  return ub.f;
}

__device__ __forceinline__ void store_strip(v8f acc, int mbase, int col, int Nc,
                                            float* Cf, unsigned short* Cb,
                                            const float* bias, int act) {
  float badd = bias ? bias[col] : 0.0f;
#pragma unroll
  for (int v = 0; v < 8; ++v) {
    float val = acc[v] + badd;
    if (act == 1) val = (val > 20.f) ? val : log1pf(__expf(val)); // softplus
    size_t o = (size_t)(mbase + v) * Nc + col;
    if (Cf) Cf[o] = val;
    if (Cb) Cb[o] = f2bf(val);
  }
}

#define WMMA_BF16(ACC, AF, BF) \
  ACC = __builtin_amdgcn_wmma_f32_16x16x32_bf16(false, AF, false, BF, (short)0, ACC, false, false)

__global__ __launch_bounds__(128)
void wmma_gemm_bf16(const __bf16* __restrict__ A, const __bf16* __restrict__ W,
                    float* __restrict__ Cf, unsigned short* __restrict__ Cb,
                    const float* __restrict__ bias,
                    int M, int Nc, int K, int act) {
  const int lane = threadIdx.x & 31;
  const int wave = threadIdx.x >> 5;
  const int row0 = (blockIdx.y * 4 + wave) * 32;   // 32 M-rows per wave
  const int col0 = blockIdx.x * 64;
  if (row0 >= M) return;
  const int r   = lane & 15;
  const int hlf = lane >> 4;

  const __bf16* aRow0 = A + (size_t)(row0 +  0 + r) * K;
  const __bf16* aRow1 = A + (size_t)(row0 + 16 + r) * K;
  const __bf16* bRow0 = W + (size_t)(col0 +  0 + r) * K;
  const __bf16* bRow1 = W + (size_t)(col0 + 16 + r) * K;
  const __bf16* bRow2 = W + (size_t)(col0 + 32 + r) * K;
  const __bf16* bRow3 = W + (size_t)(col0 + 48 + r) * K;

  v8f acc00 = {}, acc01 = {}, acc02 = {}, acc03 = {};
  v8f acc10 = {}, acc11 = {}, acc12 = {}, acc13 = {};

  // --- pipeline prologue: load k=0 fragments (ping) ---
  v16bf aP0 = load_afrag(aRow0, 0, hlf), aP1 = load_afrag(aRow1, 0, hlf);
  v16bf bP0 = load_bfrag(bRow0, 0, hlf), bP1 = load_bfrag(bRow1, 0, hlf);
  v16bf bP2 = load_bfrag(bRow2, 0, hlf), bP3 = load_bfrag(bRow3, 0, hlf);
  v16bf aQ0, aQ1, bQ0, bQ1, bQ2, bQ3;

  for (int k0 = 0; k0 < K; k0 += 64) {
    // load k0+32 (pong) while computing k0 (ping)
    aQ0 = load_afrag(aRow0, k0 + 32, hlf); aQ1 = load_afrag(aRow1, k0 + 32, hlf);
    bQ0 = load_bfrag(bRow0, k0 + 32, hlf); bQ1 = load_bfrag(bRow1, k0 + 32, hlf);
    bQ2 = load_bfrag(bRow2, k0 + 32, hlf); bQ3 = load_bfrag(bRow3, k0 + 32, hlf);
    if (k0 + 128 < K) {                      // near-cache prefetch, 256B ahead
      __builtin_prefetch(aRow0 + k0 + 128, 0, 3);
      __builtin_prefetch(aRow1 + k0 + 128, 0, 3);
      __builtin_prefetch(bRow0 + k0 + 128, 0, 3);
      __builtin_prefetch(bRow1 + k0 + 128, 0, 3);
      __builtin_prefetch(bRow2 + k0 + 128, 0, 3);
      __builtin_prefetch(bRow3 + k0 + 128, 0, 3);
    }
    WMMA_BF16(acc00, aP0, bP0); WMMA_BF16(acc01, aP0, bP1);
    WMMA_BF16(acc02, aP0, bP2); WMMA_BF16(acc03, aP0, bP3);
    WMMA_BF16(acc10, aP1, bP0); WMMA_BF16(acc11, aP1, bP1);
    WMMA_BF16(acc12, aP1, bP2); WMMA_BF16(acc13, aP1, bP3);

    // load k0+64 (ping) while computing k0+32 (pong)
    if (k0 + 64 < K) {
      aP0 = load_afrag(aRow0, k0 + 64, hlf); aP1 = load_afrag(aRow1, k0 + 64, hlf);
      bP0 = load_bfrag(bRow0, k0 + 64, hlf); bP1 = load_bfrag(bRow1, k0 + 64, hlf);
      bP2 = load_bfrag(bRow2, k0 + 64, hlf); bP3 = load_bfrag(bRow3, k0 + 64, hlf);
    }
    WMMA_BF16(acc00, aQ0, bQ0); WMMA_BF16(acc01, aQ0, bQ1);
    WMMA_BF16(acc02, aQ0, bQ2); WMMA_BF16(acc03, aQ0, bQ3);
    WMMA_BF16(acc10, aQ1, bQ0); WMMA_BF16(acc11, aQ1, bQ1);
    WMMA_BF16(acc12, aQ1, bQ2); WMMA_BF16(acc13, aQ1, bQ3);
  }

  const int mb0 = row0 + hlf * 8;
  const int mb1 = row0 + 16 + hlf * 8;
  store_strip(acc00, mb0, col0 +  0 + r, Nc, Cf, Cb, bias, act);
  store_strip(acc01, mb0, col0 + 16 + r, Nc, Cf, Cb, bias, act);
  store_strip(acc02, mb0, col0 + 32 + r, Nc, Cf, Cb, bias, act);
  store_strip(acc03, mb0, col0 + 48 + r, Nc, Cf, Cb, bias, act);
  store_strip(acc10, mb1, col0 +  0 + r, Nc, Cf, Cb, bias, act);
  store_strip(acc11, mb1, col0 + 16 + r, Nc, Cf, Cb, bias, act);
  store_strip(acc12, mb1, col0 + 32 + r, Nc, Cf, Cb, bias, act);
  store_strip(acc13, mb1, col0 + 48 + r, Nc, Cf, Cb, bias, act);
}

// ------- fused: depthwise conv(K=4)+SiLU -> selective scan -> gate -> bf16 -------
// One wave32 per (b,h) head. Lane l: p = l&15 (channel in head), n in [ (l>>4)*8, +8 ).
// Conv window kept in registers (channel p of head h maps to x column (h/4)*16+p).
// Lanes 0-15 emit the gated output directly: yg = bf16((y + D*xact) * silu(z)).
__global__ __launch_bounds__(256)
void scan_fused_kernel(const float* __restrict__ zxbc, const float* __restrict__ dt,
                       const float* __restrict__ cw, const float* __restrict__ cb,
                       const float* __restrict__ Alog, const float* __restrict__ Dv,
                       unsigned short* __restrict__ yg) {
  const int lane = threadIdx.x & 31;
  const int wave = threadIdx.x >> 5;
  const int head = blockIdx.x * 8 + wave;   // 0 .. B*H-1 = 511
  const int b = head >> 8;                  // / H_
  const int h = head & (H_ - 1);
  const int p  = lane & 15;                 // channel within head (owned row)
  const int n0 = (lane >> 4) * 8;           // state cols [n0, n0+8)

  const int dcol = h * 16 + p;              // DI-channel of this lane
  const int xcol = DI_ + (h >> 2) * 16 + p;             // conv input column
  const int bcol = DI_ + DXB_ + (h >> 2) * 16 + n0;     // B (repeated over heads)
  const int ccol = DI_ + 2 * DXB_ + h * 16 + n0;        // C per head

  float An[8];
#pragma unroll
  for (int j = 0; j < 8; ++j)
    An[j] = -__expf(Alog[(size_t)dcol * 16 + n0 + j]);

  float st[8];
#pragma unroll
  for (int j = 0; j < 8; ++j) st[j] = 0.f;

  const float w0 = cw[dcol*4+0], w1 = cw[dcol*4+1], w2 = cw[dcol*4+2], w3 = cw[dcol*4+3];
  const float bcv = cb[dcol];
  const float Dp  = Dv[dcol];

  const size_t rowZ = (size_t)b * LL * ZW;
  float x0 = 0.f, x1 = 0.f, x2 = 0.f;       // causal conv window

  for (int t = 0; t < LL; ++t) {
    const size_t zoff = rowZ + (size_t)t * ZW;
    const size_t doff = ((size_t)b * LL + t) * DI_;
    const float x3  = zxbc[zoff + xcol];    // raw x (pre-conv)
    const float dtp = dt[doff + dcol];
    const float zv  = zxbc[zoff + dcol];    // z gate input
    const float* Bp = zxbc + zoff + bcol;
    const float* Cp = zxbc + zoff + ccol;
    if (t + 1 < LL) {                       // hide next-step load latency
      __builtin_prefetch(zxbc + zoff + ZW + bcol, 0, 3);
      __builtin_prefetch(zxbc + zoff + ZW + ccol, 0, 3);
      __builtin_prefetch(zxbc + zoff + ZW + xcol, 0, 3);
      __builtin_prefetch(zxbc + zoff + ZW + dcol, 0, 3);
      __builtin_prefetch(dt + doff + DI_ + dcol, 0, 3);
    }
    // depthwise causal conv + SiLU
    const float cv = w0*x0 + w1*x1 + w2*x2 + w3*x3 + bcv;
    const float xp = cv / (1.f + __expf(-cv));
    x0 = x1; x1 = x2; x2 = x3;

    const float dx = dtp * xp;
    float ys = 0.f;
#pragma unroll
    for (int j = 0; j < 8; ++j) {
      float dA = __expf(dtp * An[j]);
      st[j] = st[j] * dA + dx * Bp[j];
      ys += st[j] * Cp[j];
    }
    ys += __shfl_xor(ys, 16, 32);           // combine the two n-halves
    if (lane < 16) {
      const float g = zv / (1.f + __expf(-zv));          // silu(z)
      yg[doff + dcol] = f2bf((ys + Dp * xp) * g);        // gated output, bf16
    }
  }
}

// ---------------- host launcher ----------------
extern "C" void kernel_launch(void* const* d_in, const int* in_sizes, int n_in,
                              void* d_out, int out_size, void* d_ws, size_t ws_size,
                              hipStream_t stream) {
  const float* hs   = (const float*)d_in[0];
  const float* winp = (const float*)d_in[1];
  const float* wdti = (const float*)d_in[2];
  const float* wdtp = (const float*)d_in[3];
  const float* dtb  = (const float*)d_in[4];
  const float* cw   = (const float*)d_in[5];
  const float* cb   = (const float*)d_in[6];
  const float* Alog = (const float*)d_in[7];
  const float* Dv   = (const float*)d_in[8];
  const float* wout = (const float*)d_in[9];
  float* out = (float*)d_out;

  char* ws = (char*)d_ws;
  size_t off = 0;
  auto alloc = [&](size_t bytes) -> void* {
    void* p = ws + off;
    off = (off + bytes + 255) & ~(size_t)255;
    return p;
  };

  unsigned short* hs_bf   = (unsigned short*)alloc((size_t)MROWS * DMOD * 2);
  unsigned short* win_bf  = (unsigned short*)alloc((size_t)ZW * DMOD * 2);
  unsigned short* wdti_bf = (unsigned short*)alloc((size_t)DTR_ * DMOD * 2);
  unsigned short* wdtp_bf = (unsigned short*)alloc((size_t)DI_ * DTR_ * 2);
  unsigned short* wout_bf = (unsigned short*)alloc((size_t)DMOD * DI_ * 2);
  float*          zxbc    = (float*)alloc((size_t)MROWS * ZW * 4);
  unsigned short* dtr_bf  = (unsigned short*)alloc((size_t)MROWS * DTR_ * 2);
  float*          dtf     = (float*)alloc((size_t)MROWS * DI_ * 4);
  unsigned short* yg      = (unsigned short*)alloc((size_t)MROWS * DI_ * 2);
  (void)ws_size; (void)in_sizes; (void)n_in; (void)out_size;

  auto cvt = [&](const float* s, unsigned short* d, int n) {
    cvt_f32_bf16<<<dim3((n + 255) / 256), dim3(256), 0, stream>>>(s, d, n);
  };
  cvt(hs,   hs_bf,   MROWS * DMOD);
  cvt(winp, win_bf,  ZW * DMOD);
  cvt(wdti, wdti_bf, DTR_ * DMOD);
  cvt(wdtp, wdtp_bf, DI_ * DTR_);
  cvt(wout, wout_bf, DMOD * DI_);

  // GEMM1: zxbc = hs * W_in^T      (4096 x 10240, K=2048)
  wmma_gemm_bf16<<<dim3(ZW / 64, MROWS / 128), dim3(128), 0, stream>>>(
      (const __bf16*)hs_bf, (const __bf16*)win_bf, zxbc, nullptr, nullptr,
      MROWS, ZW, DMOD, 0);

  // GEMM2: dtr = hs * Wdt_in^T     (4096 x 128, K=2048) -> bf16 for next GEMM
  wmma_gemm_bf16<<<dim3(DTR_ / 64, MROWS / 128), dim3(128), 0, stream>>>(
      (const __bf16*)hs_bf, (const __bf16*)wdti_bf, nullptr, dtr_bf, nullptr,
      MROWS, DTR_, DMOD, 0);

  // GEMM3: dt = softplus(dtr * Wdt^T + b)  (4096 x 4096, K=128), fused epilogue
  wmma_gemm_bf16<<<dim3(DI_ / 64, MROWS / 128), dim3(128), 0, stream>>>(
      (const __bf16*)dtr_bf, (const __bf16*)wdtp_bf, dtf, nullptr, dtb,
      MROWS, DI_, DTR_, 1);

  // fused conv+SiLU+scan+gate: 512 heads, 8 waves/block -> yg (bf16)
  scan_fused_kernel<<<dim3((BB * H_) / 8), dim3(256), 0, stream>>>(
      zxbc, dtf, cw, cb, Alog, Dv, yg);

  // GEMM4: out = yg * W_out^T      (4096 x 2048, K=4096)
  wmma_gemm_bf16<<<dim3(DMOD / 64, MROWS / 128), dim3(128), 0, stream>>>(
      (const __bf16*)yg, (const __bf16*)wout_bf, out, nullptr, nullptr,
      MROWS, DMOD, DI_, 0);
}
